// DynamicToepliztMultihead_45380624450199
// MI455X (gfx1250) — matrix operand order repaired and checked by reading, not compiled, not run on hardware
//
#include <hip/hip_runtime.h>
#include <hip/hip_bf16.h>

// Problem constants (fixed by the reference).
#define BB   16
#define HH   8
#define NN   2048
#define EE   64
#define PD   16
#define N2   (2*NN)          // 4096
#define KT   32              // K per WMMA step
#define NKT  (NN / KT)       // 64 K-steps
#define BM   128             // rows per workgroup (8 waves x 16)

typedef __attribute__((ext_vector_type(16))) __bf16 v16bf;
typedef __attribute__((ext_vector_type(8)))  float  v8f;

// ---------------------------------------------------------------------------
// Kernel 1: DynamicPosBias MLP -> reversed, exp'ed, scaled Toeplitz generator
//   arev[h*N2 + p] = bf16( exp(clip(decay(k)+dpb(k)[h], -60, 30)) / sqrt(2n) )
//   with k = (n-1-p) mod 2n, so that T[i,j] = arev[(n-1) - i + j].
// ---------------------------------------------------------------------------
__device__ __forceinline__ void ln_relu(const float* __restrict__ in,
                                        const float* __restrict__ g,
                                        const float* __restrict__ b,
                                        float* __restrict__ outv) {
    float m = 0.f;
#pragma unroll
    for (int c = 0; c < PD; ++c) m += in[c];
    m *= (1.f / PD);
    float v = 0.f;
#pragma unroll
    for (int c = 0; c < PD; ++c) { float d = in[c] - m; v += d * d; }
    v *= (1.f / PD);
    float rs = rsqrtf(v + 1e-5f);
#pragma unroll
    for (int c = 0; c < PD; ++c) {
        float y = (in[c] - m) * rs * g[c] + b[c];
        outv[c] = y > 0.f ? y : 0.f;
    }
}

__global__ void dpb_prep_kernel(const float* __restrict__ gamma,
                                const float* __restrict__ w0, const float* __restrict__ b0,
                                const float* __restrict__ g1, const float* __restrict__ gb1,
                                const float* __restrict__ w1, const float* __restrict__ b1,
                                const float* __restrict__ g2, const float* __restrict__ gb2,
                                const float* __restrict__ w2, const float* __restrict__ b2,
                                const float* __restrict__ g3, const float* __restrict__ gb3,
                                const float* __restrict__ w3, const float* __restrict__ b3,
                                __bf16* __restrict__ arev) {
    const int p = blockIdx.x * blockDim.x + threadIdx.x;
    if (p >= N2) return;

    int k = (NN - 1) - p;
    if (k < 0) k += N2;

    const float ls = -log1pf(expf(-gamma[0]));   // log(sigmoid(gamma))
    float t, decay;
    if (k == 0 || k == NN)      { t = 0.f;               decay = 0.f; }
    else if (k < NN)            { t = (float)k;          decay = ls * (float)k; }
    else                        { t = (float)(k - NN);   decay = ls * (float)(N2 - 1 - k); }

    float u[PD], r[PD];
#pragma unroll
    for (int c = 0; c < PD; ++c) u[c] = t * w0[c] + b0[c];
    ln_relu(u, g1, gb1, r);
#pragma unroll
    for (int c = 0; c < PD; ++c) {
        float s = b1[c];
#pragma unroll
        for (int q = 0; q < PD; ++q) s += r[q] * w1[q * PD + c];
        u[c] = s;
    }
    ln_relu(u, g2, gb2, r);
#pragma unroll
    for (int c = 0; c < PD; ++c) {
        float s = b2[c];
#pragma unroll
        for (int q = 0; q < PD; ++q) s += r[q] * w2[q * PD + c];
        u[c] = s;
    }
    ln_relu(u, g3, gb3, r);

    const float invsq = rsqrtf((float)N2);       // ortho-FFT scaling folded in
#pragma unroll
    for (int hh = 0; hh < HH; ++hh) {
        float o = b3[hh];
#pragma unroll
        for (int q = 0; q < PD; ++q) o += r[q] * w3[q * HH + hh];
        float val = decay + o;
        val = fminf(fmaxf(val, -60.f), 30.f);
        arev[hh * N2 + p] = (__bf16)(expf(val) * invsq);
    }
}

// ---------------------------------------------------------------------------
// Kernel 2: Toeplitz GEMM  out[b,h] = T[h] @ x[b,h]   via v_wmma_f32_16x16x32_bf16
//   Workgroup: 256 threads (8 wave32), 128x64 output tile per (b,h).
//   Double-buffered bf16 x-tile in LDS, one barrier per K-step.
// ---------------------------------------------------------------------------
__device__ __forceinline__ unsigned pack_bf16(float lo, float hi) {
    union { __bf16 h[2]; unsigned u; } p;
    p.h[0] = (__bf16)lo;
    p.h[1] = (__bf16)hi;
    return p.u;
}

__global__ void __launch_bounds__(256)
toeplitz_wmma_kernel(const float* __restrict__ x,
                     const __bf16* __restrict__ arev,
                     float* __restrict__ out) {
    __shared__ alignas(64) __bf16 aS[N2];            // 8 KB: reversed generator
    __shared__ alignas(64) __bf16 xT[2][EE][KT];     // 2 x 4 KB: transposed x K-tiles

    const int tid  = threadIdx.x;
    const int bh   = blockIdx.y;                     // b*H + h
    const int h    = bh & (HH - 1);
    const int m0   = blockIdx.x * BM;

    // Stage this head's generator into LDS (4096 bf16 = 512 uint4).
    {
        const uint4* src = (const uint4*)(arev + (size_t)h * N2);
        uint4* dst = (uint4*)aS;
#pragma unroll
        for (int i = tid; i < N2 / 8; i += 256) dst[i] = src[i];
    }

    const size_t xbase = (size_t)bh * NN * EE;
    const int lane = tid & 31;
    const int wave = tid >> 5;                       // 0..7
    const int lr   = lane & 15;
    const int hi   = lane >> 4;                      // lane group 0/1
    const int mrow = m0 + wave * 16 + lr;            // A-matrix row of this lane
    const int cA   = (NN - 1) - mrow;                // T[i,j] = aS[cA + j]
    const int krow = hi ? 16 : 0;                    // B-fragment K group

    // Staging map: thread owns K-row pair (2*kp, 2*kp+1) x 4 columns [e0s, e0s+4).
    const int kp  = tid >> 4;                        // 0..15
    const int e0s = (tid & 15) << 2;                 // 0,4,...,60
    const float* __restrict__ xrow =
        x + xbase + (size_t)(2 * kp) * EE + e0s;     // row 2*kp of tile 0

    v8f acc0 = {0.f,0.f,0.f,0.f,0.f,0.f,0.f,0.f};
    v8f acc1 = acc0, acc2 = acc0, acc3 = acc0;

    // ---- prologue: stage tile 0 into buffer 0 ----
    {
        float4 r0 = *(const float4*)(xrow);
        float4 r1 = *(const float4*)(xrow + EE);
        unsigned* d0 = (unsigned*)&xT[0][e0s + 0][2 * kp];
        unsigned* d1 = (unsigned*)&xT[0][e0s + 1][2 * kp];
        unsigned* d2 = (unsigned*)&xT[0][e0s + 2][2 * kp];
        unsigned* d3 = (unsigned*)&xT[0][e0s + 3][2 * kp];
        *d0 = pack_bf16(r0.x, r1.x);
        *d1 = pack_bf16(r0.y, r1.y);
        *d2 = pack_bf16(r0.z, r1.z);
        *d3 = pack_bf16(r0.w, r1.w);
    }
    __syncthreads();

    for (int kt = 0; kt < NKT; ++kt) {
        const int cur = kt & 1;
        const int nxt = cur ^ 1;
        const bool more = (kt + 1 < NKT);

        // Issue global loads for the next tile early (hide latency under WMMA).
        float4 r0, r1;
        if (more) {
            const float* p = xrow + (size_t)(kt + 1) * (KT * EE);
            r0 = *(const float4*)(p);
            r1 = *(const float4*)(p + EE);
        }

        // A fragment (16x32 bf16): lane<16 holds K 0-15; lane>=16 K 16-31.
        const int b1i = cA + kt * KT + (hi ? 16 : 0);
        const int b2i = b1i + 8;
        v16bf afrag;
#pragma unroll
        for (int i = 0; i < 8; ++i) afrag[i]     = aS[b1i + i];
#pragma unroll
        for (int i = 0; i < 8; ++i) afrag[8 + i] = aS[b2i + i];

        // B fragments: lanes 0-15 -> K 0-15, lanes 16-31 -> K 16-31, contiguous in LDS.
        const v16bf bf0 = *(const v16bf*)&xT[cur][ 0 + lr][krow];
        const v16bf bf1 = *(const v16bf*)&xT[cur][16 + lr][krow];
        const v16bf bf2 = *(const v16bf*)&xT[cur][32 + lr][krow];
        const v16bf bf3 = *(const v16bf*)&xT[cur][48 + lr][krow];

        acc0 = __builtin_amdgcn_wmma_f32_16x16x32_bf16(false, afrag, false, bf0, (short)0, acc0, false, false);
        acc1 = __builtin_amdgcn_wmma_f32_16x16x32_bf16(false, afrag, false, bf1, (short)0, acc1, false, false);
        acc2 = __builtin_amdgcn_wmma_f32_16x16x32_bf16(false, afrag, false, bf2, (short)0, acc2, false, false);
        acc3 = __builtin_amdgcn_wmma_f32_16x16x32_bf16(false, afrag, false, bf3, (short)0, acc3, false, false);

        // Convert + store next tile into the other buffer (nobody reads it now).
        if (more) {
            unsigned* d0 = (unsigned*)&xT[nxt][e0s + 0][2 * kp];
            unsigned* d1 = (unsigned*)&xT[nxt][e0s + 1][2 * kp];
            unsigned* d2 = (unsigned*)&xT[nxt][e0s + 2][2 * kp];
            unsigned* d3 = (unsigned*)&xT[nxt][e0s + 3][2 * kp];
            *d0 = pack_bf16(r0.x, r1.x);
            *d1 = pack_bf16(r0.y, r1.y);
            *d2 = pack_bf16(r0.z, r1.z);
            *d3 = pack_bf16(r0.w, r1.w);
        }
        __syncthreads();
    }

    // D layout: VGPR r -> row (wave*16 + hi*8 + r), column lr in each 16-col group.
    float* __restrict__ o = out + xbase + (size_t)(m0 + wave * 16 + hi * 8) * EE;
#pragma unroll
    for (int r = 0; r < 8; ++r) {
        __builtin_nontemporal_store(acc0[r], &o[(size_t)r * EE +  0 + lr]);
        __builtin_nontemporal_store(acc1[r], &o[(size_t)r * EE + 16 + lr]);
        __builtin_nontemporal_store(acc2[r], &o[(size_t)r * EE + 32 + lr]);
        __builtin_nontemporal_store(acc3[r], &o[(size_t)r * EE + 48 + lr]);
    }
}

// ---------------------------------------------------------------------------
extern "C" void kernel_launch(void* const* d_in, const int* in_sizes, int n_in,
                              void* d_out, int out_size, void* d_ws, size_t ws_size,
                              hipStream_t stream) {
    (void)in_sizes; (void)n_in; (void)out_size; (void)ws_size;
    const float* x     = (const float*)d_in[0];
    const float* gamma = (const float*)d_in[1];
    const float* w0    = (const float*)d_in[2];
    const float* b0    = (const float*)d_in[3];
    const float* g1    = (const float*)d_in[4];
    const float* gb1   = (const float*)d_in[5];
    const float* w1    = (const float*)d_in[6];
    const float* b1    = (const float*)d_in[7];
    const float* g2    = (const float*)d_in[8];
    const float* gb2   = (const float*)d_in[9];
    const float* w2    = (const float*)d_in[10];
    const float* b2    = (const float*)d_in[11];
    const float* g3    = (const float*)d_in[12];
    const float* gb3   = (const float*)d_in[13];
    const float* w3    = (const float*)d_in[14];
    const float* b3    = (const float*)d_in[15];
    float* out = (float*)d_out;

    __bf16* arev = (__bf16*)d_ws;                 // HH * N2 bf16 = 64 KB

    dpb_prep_kernel<<<dim3(N2 / 256), dim3(256), 0, stream>>>(
        gamma, w0, b0, g1, gb1, w1, b1, g2, gb2, w2, b2, g3, gb3, w3, b3, arev);

    toeplitz_wmma_kernel<<<dim3(NN / BM, BB * HH), dim3(256), 0, stream>>>(
        x, arev, out);
}
